// TaskResourceGNN_5875515261459
// MI455X (gfx1250) — matrix operand order repaired and checked by reading, not compiled, not run on hardware
//
#include <hip/hip_runtime.h>
#include <float.h>

typedef __attribute__((ext_vector_type(16))) __bf16 v16bf;
typedef __attribute__((ext_vector_type(8)))  __bf16 v8bf;
typedef __attribute__((ext_vector_type(8)))  float  v8f;

#define LRELU_SLOPE 0.2f
#define BN_INVS 0.9999950000374997f   /* 1/sqrt(1+1e-5) */

static __device__ __forceinline__ unsigned short f2bf(float f) {
  unsigned int u = __float_as_uint(f);
  unsigned int lsb = (u >> 16) & 1u;
  u += 0x7fffu + lsb;                 /* round to nearest even */
  return (unsigned short)(u >> 16);
}

static __device__ __forceinline__ void atomicMaxF(float* addr, float v) {
  if (v >= 0.0f) atomicMax((int*)addr, __float_as_int(v));
  else           atomicMin((unsigned int*)addr, __float_as_uint(v));
}

/* build A operand (16x32 bf16 tile): lane group g holds K[16g..16g+15] */
static __device__ __forceinline__ v16bf loadA(const unsigned short* p, int g) {
  v8bf lo = *(const v8bf*)(p + 16 * g);
  v8bf hi = *(const v8bf*)(p + 16 * g + 8);
  v16bf r;
#pragma unroll
  for (int i = 0; i < 8; ++i) { r[i] = lo[i]; r[8 + i] = hi[i]; }
  return r;
}

/* ---------------- fill ---------------- */
__global__ void k_fill(float* __restrict__ p, float v, int n) {
  int i = blockIdx.x * blockDim.x + threadIdx.x;
  if (i < n) p[i] = v;
}

/* ---- weight prep: W[K,Nout] f32 -> Wt[Nout,K] bf16 (B operand layout) ---- */
__global__ void k_transw(const float* __restrict__ W, unsigned short* __restrict__ Wt,
                         int K, int Nout) {
  int i = blockIdx.x * blockDim.x + threadIdx.x;
  if (i >= K * Nout) return;
  int n = i / K, k = i - n * K;
  Wt[i] = f2bf(W[(size_t)k * Nout + n]);
}

/* ---- node encoder: relu(x@W+b) -> bf16, ld 128 ---- */
__global__ void k_node_enc(const float* __restrict__ x, const float* __restrict__ W,
                           const float* __restrict__ b, unsigned short* __restrict__ xbf) {
  int n = blockIdx.x, j = threadIdx.x;
  float a = b[j];
#pragma unroll
  for (int t = 0; t < 10; ++t) a += x[n * 10 + t] * W[t * 128 + j];
  a = fmaxf(a, 0.0f);
  xbf[(size_t)n * 128 + j] = f2bf(a);
}

/* ---- platform encoder: relu(px@W+b) -> bf16[64] ---- */
__global__ void k_plat_enc(const float* __restrict__ px, const float* __restrict__ W,
                           const float* __restrict__ b, unsigned short* __restrict__ pbf) {
  int j = threadIdx.x;
  float a = b[j];
#pragma unroll
  for (int t = 0; t < 4; ++t) a += px[t] * W[t * 64 + j];
  a = fmaxf(a, 0.0f);
  pbf[j] = f2bf(a);
}

/* ---- bf16 WMMA GEMM: C[M,Nout] = A[M,K] * Bt[Nout,K]^T
   One wave handles TWO stacked 16x16 M-tiles (shared B) and splits K into
   even/odd 32-chunks -> 4 independent WMMA accumulator chains for XDL ILP.
   blockDim.x = Nout*2, gridDim.x = ceil(M/32). K must be a multiple of 64. ---- */
__global__ __launch_bounds__(256)
void k_wmma_gemm(const unsigned short* __restrict__ A, int lda,
                 const unsigned short* __restrict__ Bt,
                 float* __restrict__ C, int ldc, int M, int K) {
  const int lane = threadIdx.x & 31;
  const int wv   = threadIdx.x >> 5;
  const int g    = lane >> 4;
  const int l15  = lane & 15;
  int mrow0 = blockIdx.x * 32 + l15;
  int mrow1 = blockIdx.x * 32 + 16 + l15;
  if (mrow0 >= M) mrow0 = M - 1;          /* keep EXEC all-ones for WMMA */
  if (mrow1 >= M) mrow1 = M - 1;
  const int ncol = wv * 16 + l15;
  const unsigned short* a0 = A + (size_t)mrow0 * lda;
  const unsigned short* a1 = A + (size_t)mrow1 * lda;
  const unsigned short* bPtr = Bt + (size_t)ncol * K + 16 * g;
  v8f acc00 = {}, acc01 = {}, acc10 = {}, acc11 = {};
  for (int k0 = 0; k0 < K; k0 += 64) {
    v16bf av00 = loadA(a0 + k0, g);
    v16bf av01 = loadA(a0 + k0 + 32, g);
    v16bf av10 = loadA(a1 + k0, g);
    v16bf av11 = loadA(a1 + k0 + 32, g);
    v16bf bv0 = *(const v16bf*)(bPtr + k0);
    v16bf bv1 = *(const v16bf*)(bPtr + k0 + 32);
    acc00 = __builtin_amdgcn_wmma_f32_16x16x32_bf16(false, av00, false, bv0,
                                                    (short)0, acc00, false, false);
    acc10 = __builtin_amdgcn_wmma_f32_16x16x32_bf16(false, av10, false, bv0,
                                                    (short)0, acc10, false, false);
    acc01 = __builtin_amdgcn_wmma_f32_16x16x32_bf16(false, av01, false, bv1,
                                                    (short)0, acc01, false, false);
    acc11 = __builtin_amdgcn_wmma_f32_16x16x32_bf16(false, av11, false, bv1,
                                                    (short)0, acc11, false, false);
  }
  v8f acc0 = acc00 + acc01;
  v8f acc1 = acc10 + acc11;
  const int rowBase0 = blockIdx.x * 32 + g * 8;
  const int rowBase1 = blockIdx.x * 32 + 16 + g * 8;
#pragma unroll
  for (int v = 0; v < 8; ++v) {
    int r = rowBase0 + v;
    if (r < M) C[(size_t)r * ldc + ncol] = acc0[v];
  }
#pragma unroll
  for (int v = 0; v < 8; ++v) {
    int r = rowBase1 + v;
    if (r < M) C[(size_t)r * ldc + ncol] = acc1[v];
  }
}

/* ---- attention coefficients: als/ald[n,h] = sum_d h[n,h,d]*a[h,d] ---- */
__global__ void k_attn_coef(const float* __restrict__ h, const float* __restrict__ asrc,
                            const float* __restrict__ adst, float* __restrict__ als,
                            float* __restrict__ ald, int heads) {
  __shared__ float ps[128], pd[128];
  const int n = blockIdx.x, t = threadIdx.x;
  float v = h[(size_t)n * 128 + t];
  ps[t] = v * asrc[t];
  pd[t] = v * adst[t];
  __syncthreads();
  const int dph = 128 / heads;            /* 32 or 128, power of 2 */
  for (int off = dph >> 1; off > 0; off >>= 1) {
    if ((t & (dph - 1)) < off) { ps[t] += ps[t + off]; pd[t] += pd[t + off]; }
    __syncthreads();
  }
  if ((t & (dph - 1)) == 0) {
    int hd = t / dph;
    als[(size_t)n * heads + hd] = ps[t];
    ald[(size_t)n * heads + hd] = pd[t];
  }
}

/* ---- pass 1: segment max over destinations ---- */
__global__ void k_edge_max(const int* __restrict__ src, const int* __restrict__ dst,
                           const float* __restrict__ als, const float* __restrict__ ald,
                           float* __restrict__ m, int E, int EA, int heads) {
  int e = blockIdx.x * blockDim.x + threadIdx.x;
  if (e >= EA) return;
  __builtin_prefetch(src + e + 2048, 0, 0);
  int s, d;
  if (e < E) { s = src[e]; d = dst[e]; } else { s = d = e - E; }
  for (int hh = 0; hh < heads; ++hh) {
    float v = als[(size_t)s * heads + hh] + ald[(size_t)d * heads + hh];
    v = (v > 0.0f) ? v : LRELU_SLOPE * v;
    atomicMaxF(&m[(size_t)d * heads + hh], v);
  }
}

/* ---- pass 2: segment sum of exp(e - m[dst]) ---- */
__global__ void k_edge_expsum(const int* __restrict__ src, const int* __restrict__ dst,
                              const float* __restrict__ als, const float* __restrict__ ald,
                              const float* __restrict__ m, float* __restrict__ ssum,
                              int E, int EA, int heads) {
  int e = blockIdx.x * blockDim.x + threadIdx.x;
  if (e >= EA) return;
  __builtin_prefetch(dst + e + 2048, 0, 0);
  int s, d;
  if (e < E) { s = src[e]; d = dst[e]; } else { s = d = e - E; }
  for (int hh = 0; hh < heads; ++hh) {
    float v = als[(size_t)s * heads + hh] + ald[(size_t)d * heads + hh];
    v = (v > 0.0f) ? v : LRELU_SLOPE * v;
    float ex = __expf(v - m[(size_t)d * heads + hh]);
    atomicAdd(&ssum[(size_t)d * heads + hh], ex);
  }
}

/* ---- pass 3: acc[dst] += alpha * h[src]; one wave32 per edge ---- */
__global__ void k_edge_agg(const int* __restrict__ src, const int* __restrict__ dst,
                           const float* __restrict__ als, const float* __restrict__ ald,
                           const float* __restrict__ m, const float* __restrict__ ssum,
                           const float* __restrict__ h, float* __restrict__ acc,
                           int E, int EA, int heads, int dimShift) {
  int gw = (blockIdx.x * blockDim.x + threadIdx.x) >> 5;
  int lane = threadIdx.x & 31;
  if (gw >= EA) return;
  int s, d;
  if (gw < E) { s = src[gw]; d = dst[gw]; } else { s = d = gw - E; }
  int hd = (lane * 4) >> dimShift;                 /* 5 -> heads=4, 7 -> heads=1 */
  float e = als[(size_t)s * heads + hd] + ald[(size_t)d * heads + hd];
  e = (e > 0.0f) ? e : LRELU_SLOPE * e;
  float alpha = __expf(e - m[(size_t)d * heads + hd]) /
                (ssum[(size_t)d * heads + hd] + 1e-16f);
  const float4 hv = *(const float4*)(h + (size_t)s * 128 + lane * 4);
  float* o = acc + (size_t)d * 128 + lane * 4;
  atomicAdd(o + 0, alpha * hv.x);
  atomicAdd(o + 1, alpha * hv.y);
  atomicAdd(o + 2, alpha * hv.z);
  atomicAdd(o + 3, alpha * hv.w);
}

/* ---- epilogue: + bias, optional eval-BN, ReLU; optional f32 and bf16 out ---- */
__global__ void k_epilogue(const float* __restrict__ acc, const float* __restrict__ bias,
                           const float* __restrict__ bng, const float* __restrict__ bnb,
                           float* __restrict__ outf, unsigned short* __restrict__ outbf,
                           int ldbf, int ncols, int total) {
  int i = blockIdx.x * blockDim.x + threadIdx.x;
  if (i >= total) return;
  int r = i / ncols, c = i - r * ncols;
  float v = acc[i] + bias[c];
  if (bng) v = v * (bng[c] * BN_INVS) + bnb[c];
  v = fmaxf(v, 0.0f);
  if (outf) outf[i] = v;
  if (outbf) outbf[(size_t)r * ldbf + c] = f2bf(v);
}

/* ---- broadcast platform embedding into cols 128..191 of xbf (ld 192) ---- */
__global__ void k_bcast_p(const unsigned short* __restrict__ pbf,
                          unsigned short* __restrict__ xbf, int total) {
  int i = blockIdx.x * blockDim.x + threadIdx.x;
  if (i >= total) return;
  int n = i >> 6, j = i & 63;
  xbf[(size_t)n * 192 + 128 + j] = pbf[j];
}

/* ---- final tiny GEMV: [N,64] @ [64,3] + b ---- */
__global__ void k_c3(const float* __restrict__ hin, const float* __restrict__ W,
                     const float* __restrict__ b, float* __restrict__ out, int N) {
  int n = blockIdx.x * blockDim.x + threadIdx.x;
  if (n >= N) return;
  float a0 = b[0], a1 = b[1], a2 = b[2];
#pragma unroll 8
  for (int t = 0; t < 64; ++t) {
    float v = hin[(size_t)n * 64 + t];
    a0 += v * W[t * 3 + 0];
    a1 += v * W[t * 3 + 1];
    a2 += v * W[t * 3 + 2];
  }
  out[(size_t)n * 3 + 0] = a0;
  out[(size_t)n * 3 + 1] = a1;
  out[(size_t)n * 3 + 2] = a2;
}

extern "C" void kernel_launch(void* const* d_in, const int* in_sizes, int n_in,
                              void* d_out, int out_size, void* d_ws, size_t ws_size,
                              hipStream_t stream) {
  (void)n_in; (void)out_size; (void)ws_size;
  const float* x      = (const float*)d_in[0];
  const int*   eidx   = (const int*)d_in[1];
  const float* px     = (const float*)d_in[2];
  const float* node_W = (const float*)d_in[3];
  const float* node_b = (const float*)d_in[4];
  const float* plat_W = (const float*)d_in[5];
  const float* plat_b = (const float*)d_in[6];
  const float* gW[3]  = {(const float*)d_in[7],  (const float*)d_in[11], (const float*)d_in[15]};
  const float* gas[3] = {(const float*)d_in[8],  (const float*)d_in[12], (const float*)d_in[16]};
  const float* gad[3] = {(const float*)d_in[9],  (const float*)d_in[13], (const float*)d_in[17]};
  const float* gb[3]  = {(const float*)d_in[10], (const float*)d_in[14], (const float*)d_in[18]};
  const float* bng[3] = {(const float*)d_in[19], (const float*)d_in[21], (const float*)d_in[23]};
  const float* bnb[3] = {(const float*)d_in[20], (const float*)d_in[22], (const float*)d_in[24]};
  const float* c1W = (const float*)d_in[25]; const float* c1b = (const float*)d_in[26];
  const float* c2W = (const float*)d_in[27]; const float* c2b = (const float*)d_in[28];
  const float* c3W = (const float*)d_in[29]; const float* c3b = (const float*)d_in[30];
  float* out = (float*)d_out;

  const int N  = in_sizes[0] / 10;
  const int E  = in_sizes[1] / 2;
  const int EA = E + N;
  const int* src = eidx;
  const int* dst = eidx + E;

  char* w = (char*)d_ws;
  size_t off = 0;
  auto take = [&](size_t bytes) -> char* {
    char* p = w + off; off = (off + bytes + 255) & ~(size_t)255; return p;
  };
  float* hbuf = (float*)take((size_t)N * 128 * 4);
  float* accb = (float*)take((size_t)N * 128 * 4);
  unsigned short* xbf = (unsigned short*)take((size_t)N * 192 * 2);
  float* als  = (float*)take((size_t)N * 4 * 4);
  float* ald  = (float*)take((size_t)N * 4 * 4);
  float* mbuf = (float*)take((size_t)N * 4 * 4);
  float* sbuf = (float*)take((size_t)N * 4 * 4);
  unsigned short* gt[3];
  gt[0] = (unsigned short*)take(128 * 128 * 2);
  gt[1] = (unsigned short*)take(128 * 128 * 2);
  gt[2] = (unsigned short*)take(128 * 128 * 2);
  unsigned short* c1t = (unsigned short*)take(192 * 128 * 2);
  unsigned short* c2t = (unsigned short*)take(128 * 64 * 2);
  unsigned short* pbf = (unsigned short*)take(64 * 2);

  /* weight prep (bf16, transposed to B-operand layout) */
  for (int l = 0; l < 3; ++l)
    k_transw<<<(128 * 128 + 255) / 256, 256, 0, stream>>>(gW[l], gt[l], 128, 128);
  k_transw<<<(192 * 128 + 255) / 256, 256, 0, stream>>>(c1W, c1t, 192, 128);
  k_transw<<<(128 * 64 + 255) / 256, 256, 0, stream>>>(c2W, c2t, 128, 64);
  k_plat_enc<<<1, 64, 0, stream>>>(px, plat_W, plat_b, pbf);
  k_node_enc<<<N, 128, 0, stream>>>(x, node_W, node_b, xbf);

  const int mt = (N + 31) / 32;   /* two M-tiles per wave */
  for (int l = 0; l < 3; ++l) {
    const int heads = (l == 2) ? 1 : 4;
    const int dimShift = (l == 2) ? 7 : 5;
    k_wmma_gemm<<<mt, 256, 0, stream>>>(xbf, 128, gt[l], hbuf, 128, N, 128);
    k_attn_coef<<<N, 128, 0, stream>>>(hbuf, gas[l], gad[l], als, ald, heads);
    k_fill<<<(N * heads + 255) / 256, 256, 0, stream>>>(mbuf, -FLT_MAX, N * heads);
    k_fill<<<(N * heads + 255) / 256, 256, 0, stream>>>(sbuf, 0.0f, N * heads);
    k_fill<<<(N * 128 + 255) / 256, 256, 0, stream>>>(accb, 0.0f, N * 128);
    k_edge_max<<<(EA + 255) / 256, 256, 0, stream>>>(src, dst, als, ald, mbuf, E, EA, heads);
    k_edge_expsum<<<(EA + 255) / 256, 256, 0, stream>>>(src, dst, als, ald, mbuf, sbuf, E, EA, heads);
    k_edge_agg<<<(EA + 7) / 8, 256, 0, stream>>>(src, dst, als, ald, mbuf, sbuf, hbuf, accb,
                                                 E, EA, heads, dimShift);
    const int ldo = (l == 2) ? 192 : 128;
    k_epilogue<<<(N * 128 + 255) / 256, 256, 0, stream>>>(
        accb, gb[l], bng[l], bnb[l], nullptr, xbf, ldo, 128, N * 128);
  }
  /* concat platform embedding, then classifier */
  k_bcast_p<<<(N * 64 + 255) / 256, 256, 0, stream>>>(pbf, xbf, N * 64);
  k_wmma_gemm<<<mt, 256, 0, stream>>>(xbf, 192, c1t, hbuf, 128, N, 192);
  k_epilogue<<<(N * 128 + 255) / 256, 256, 0, stream>>>(
      hbuf, c1b, nullptr, nullptr, nullptr, xbf, 128, 128, N * 128);
  k_wmma_gemm<<<mt, 128, 0, stream>>>(xbf, 128, c2t, accb, 64, N, 128);
  k_epilogue<<<(N * 64 + 255) / 256, 256, 0, stream>>>(
      accb, c2b, nullptr, nullptr, accb, nullptr, 0, 64, N * 64);
  k_c3<<<(N + 127) / 128, 128, 0, stream>>>(accb, c3W, c3b, out, N);
}
